// ScaledDotProductAttention_44203803411005
// MI455X (gfx1250) — matrix operand order repaired and checked by reading, not compiled, not run on hardware
//
#include <hip/hip_runtime.h>

#define B_ 2
#define H_ 16
#define S_ 2048
#define D_ 128
#define NTOT ((size_t)B_ * H_ * S_ * D_)

typedef __bf16 v16bf __attribute__((ext_vector_type(16)));
typedef __bf16 v8bf  __attribute__((ext_vector_type(8)));
typedef float  v8f   __attribute__((ext_vector_type(8)));
typedef unsigned uint32x4 __attribute__((ext_vector_type(4)));
typedef int      int32x8  __attribute__((ext_vector_type(8)));
typedef int      int32x4  __attribute__((ext_vector_type(4)));

#define KPITCH 136   // bf16/row (128 + 8 pad) -> 272B rows (256B data + 16B TDM pad)
#define VPITCH 40    // bf16/row (32 keys + 8 pad) -> 80B rows
#define PPITCH 40

__device__ __forceinline__ __bf16 f2bf(float f) {
  union { float f; unsigned u; } in; in.f = f;
  unsigned u = in.u;
  unsigned r = (u + 0x7FFFu + ((u >> 16) & 1u)) >> 16;
  union { unsigned short s; __bf16 b; } out; out.s = (unsigned short)r;
  return out.b;
}

__device__ __forceinline__ v16bf cat16(v8bf lo, v8bf hi) {
  return __builtin_shufflevector(lo, hi, 0,1,2,3,4,5,6,7,8,9,10,11,12,13,14,15);
}

__global__ __launch_bounds__(256)
void cvt_bf16_kernel(const float* __restrict__ s, __bf16* __restrict__ d, float scale) {
  const size_t i = ((size_t)blockIdx.x * 256 + threadIdx.x) * 8;
  float t[8];
  *(float4*)(t)     = *(const float4*)(s + i);
  *(float4*)(t + 4) = *(const float4*)(s + i + 4);
  v8bf o;
  #pragma unroll
  for (int j = 0; j < 8; ++j) o[j] = f2bf(t[j] * scale);
  *(v8bf*)(d + i) = o;
}

template <bool MASK>
__device__ __forceinline__ void score_tiles(const __bf16* Kshp, const v16bf aq[4],
                                            int j0, int qw, int col, int half,
                                            float sc[2][8]) {
  #pragma unroll
  for (int nt = 0; nt < 2; ++nt) {
    v16bf bk[4];
    #pragma unroll
    for (int c = 0; c < 4; ++c) {
      const __bf16* kp = Kshp + (nt * 16 + col) * KPITCH + 32 * c + 8 * half;
      bk[c] = cat16(*(const v8bf*)kp, *(const v8bf*)(kp + 16));
    }
    v8f s = {0.f,0.f,0.f,0.f,0.f,0.f,0.f,0.f};
    #pragma unroll
    for (int c = 0; c < 4; ++c)
      s = __builtin_amdgcn_wmma_f32_16x16x32_bf16(false, aq[c], false, bk[c],
                                                  (short)0, s, false, false);
    #pragma unroll
    for (int r = 0; r < 8; ++r) {
      if (MASK) {
        const int kg = j0 + nt * 16 + col;
        const int qg = qw + r + 8 * half;
        sc[nt][r] = (kg <= qg) ? s[r] : -1.0e9f;
      } else {
        sc[nt][r] = s[r];
      }
    }
  }
}

__global__ __launch_bounds__(256)
void fa_fwd_cdna5_kernel(const __bf16* __restrict__ Qb, const __bf16* __restrict__ Kb,
                         const __bf16* __restrict__ Vb, float* __restrict__ O) {
  __shared__ __align__(16) __bf16 Ksh[32 * KPITCH];      // K block [key][d], TDM-filled
  __shared__ __align__(16) __bf16 Vts[D_ * VPITCH];      // V block transposed [d][key]
  __shared__ __align__(16) __bf16 Psh[8 * 16 * PPITCH];  // per-wave P scratch

  const int bh   = blockIdx.y;
  const int q0   = blockIdx.x * 128;
  const int tid  = threadIdx.x;
  const int wave = tid >> 5;
  const int lane = tid & 31;
  const int col  = lane & 15;
  const int half = lane >> 4;
  const int qw   = q0 + wave * 16;

  const size_t bhOff = (size_t)bh * S_ * D_;

  // ---- Q tile -> A-layout registers (bf16, pre-scaled) ----
  v16bf aq[4];
  {
    const __bf16* qrow = Qb + bhOff + (size_t)(qw + col) * D_;
    #pragma unroll
    for (int c = 0; c < 4; ++c) {
      const int d0 = 32 * c + 8 * half;
      aq[c] = cat16(*(const v8bf*)(qrow + d0), *(const v8bf*)(qrow + d0 + 16));
    }
  }

  v8f acc[8];
  #pragma unroll
  for (int t = 0; t < 8; ++t) acc[t] = (v8f){0.f,0.f,0.f,0.f,0.f,0.f,0.f,0.f};
  float mrow[8], lrow[8];
  #pragma unroll
  for (int r = 0; r < 8; ++r) { mrow[r] = -3.0e38f; lrow[r] = 0.f; }

  const int kend = q0 + 128;
  for (int j0 = 0; j0 < kend; j0 += 32) {
    // ---- K tile: Tensor Data Mover, one descriptor per block-iteration ----
    if (wave == 0) {
      const unsigned long long ga =
          (unsigned long long)(size_t)(Kb + bhOff + (size_t)j0 * D_);
      uint32x4 g0;
      g0[0] = 1u;                                   // count=1, user mode
      g0[1] = (unsigned)(size_t)&Ksh[0];            // lds_addr
      g0[2] = (unsigned)(ga & 0xFFFFFFFFu);         // global_addr[31:0]
      g0[3] = (unsigned)((ga >> 32) & 0x1FFFFFFu)   // global_addr[56:32]
              | (2u << 30);                         // type=2 (image/tensor)
      int32x8 g1;
      g1[0] = (1 << 16)     // data_size = 2 bytes
            | (1 << 20)     // pad_enable
            | (5 << 22)     // pad_interval: every 64 DWORDs (256B)
            | (3 << 25);    // pad_amount: 4 DWORDs (16B) -> 272B LDS rows
      g1[1] = (128 << 16);  // tensor_dim0 = 128 elements
      g1[2] = (32 << 16);   // tensor_dim1 = 32 rows
      g1[3] = (128 << 16);  // tile_dim0 = 128
      g1[4] = 32;           // tile_dim1 = 32, tile_dim2 = 0
      g1[5] = 128;          // tensor_dim0_stride = 128 elements
      g1[6] = 0; g1[7] = 0;
      int32x4 gz4 = {0, 0, 0, 0};
      int32x8 gz8 = {0, 0, 0, 0, 0, 0, 0, 0};
      __builtin_amdgcn_tensor_load_to_lds(g0, g1, gz4, gz4, gz8, 0);
      __builtin_amdgcn_s_wait_tensorcnt(0);
    }
    // ---- next K tile prefetch: one cacheline (=row) per lane of wave 1 ----
    if (wave == 1 && j0 + 32 < kend)
      __builtin_prefetch(Kb + bhOff + (size_t)(j0 + 32 + lane) * D_, 0, 1);
    // ---- V tile: register transpose, key-pairs packed into b32 LDS stores ----
    {
      const int key0 = (tid & 15) * 2;
      const int d0   = (tid >> 4) * 8;
      const __bf16* v0 = Vb + bhOff + (size_t)(j0 + key0) * D_ + d0;
      v8bf a = *(const v8bf*)v0;
      v8bf b = *(const v8bf*)(v0 + D_);
      #pragma unroll
      for (int j = 0; j < 8; ++j) {
        union { __bf16 h[2]; unsigned u; } p;
        p.h[0] = a[j]; p.h[1] = b[j];
        *(unsigned*)&Vts[(d0 + j) * VPITCH + key0] = p.u;
      }
      if (j0 + 32 < kend) __builtin_prefetch(v0 + 32 * D_, 0, 1);
    }
    __syncthreads();

    if (j0 <= qw + 15) {
      float sc[2][8];
      if (j0 + 31 <= qw) score_tiles<false>(Ksh, aq, j0, qw, col, half, sc);
      else               score_tiles<true >(Ksh, aq, j0, qw, col, half, sc);

      // ---- online softmax ----
      #pragma unroll
      for (int r = 0; r < 8; ++r) {
        float mb = fmaxf(sc[0][r], sc[1][r]);
        mb = fmaxf(mb, __shfl_xor(mb, 1, 32));
        mb = fmaxf(mb, __shfl_xor(mb, 2, 32));
        mb = fmaxf(mb, __shfl_xor(mb, 4, 32));
        mb = fmaxf(mb, __shfl_xor(mb, 8, 32));
        const float mn = fmaxf(mrow[r], mb);
        const float al = __expf(mrow[r] - mn);
        const float p0 = __expf(sc[0][r] - mn);
        const float p1 = __expf(sc[1][r] - mn);
        float ls = p0 + p1;
        ls += __shfl_xor(ls, 1, 32);
        ls += __shfl_xor(ls, 2, 32);
        ls += __shfl_xor(ls, 4, 32);
        ls += __shfl_xor(ls, 8, 32);
        lrow[r] = lrow[r] * al + ls;
        mrow[r] = mn;
        sc[0][r] = p0; sc[1][r] = p1;
        #pragma unroll
        for (int t = 0; t < 8; ++t) acc[t][r] *= al;
      }
      // ---- C-layout P -> LDS -> A-layout ----
      __bf16* pw = &Psh[wave * 16 * PPITCH];
      #pragma unroll
      for (int nt = 0; nt < 2; ++nt)
        #pragma unroll
        for (int r = 0; r < 8; ++r)
          pw[(r + 8 * half) * PPITCH + nt * 16 + col] = f2bf(sc[nt][r]);
      asm volatile("s_wait_dscnt 0x0" ::: "memory");
      const __bf16* prow = pw + col * PPITCH;
      v16bf ap = cat16(*(const v8bf*)(prow + 8 * half),
                       *(const v8bf*)(prow + 16 + 8 * half));
      // ---- O += P . V ----
      #pragma unroll
      for (int g = 0; g < 2; ++g) {
        v16bf bv[4];
        #pragma unroll
        for (int u = 0; u < 4; ++u) {
          const __bf16* vp = &Vts[((g * 4 + u) * 16 + col) * VPITCH + 8 * half];
          bv[u] = cat16(*(const v8bf*)vp, *(const v8bf*)(vp + 16));
        }
        #pragma unroll
        for (int u = 0; u < 4; ++u)
          acc[g * 4 + u] = __builtin_amdgcn_wmma_f32_16x16x32_bf16(
              false, ap, false, bv[u], (short)0, acc[g * 4 + u], false, false);
      }
    }
    __syncthreads();
  }

  // ---- epilogue ----
  #pragma unroll
  for (int r = 0; r < 8; ++r) {
    const float inv = 1.0f / lrow[r];
    float* orow = O + bhOff + (size_t)(qw + r + 8 * half) * D_;
    #pragma unroll
    for (int t = 0; t < 8; ++t)
      orow[t * 16 + col] = acc[t][r] * inv;
  }
}

extern "C" void kernel_launch(void* const* d_in, const int* in_sizes, int n_in,
                              void* d_out, int out_size, void* d_ws, size_t ws_size,
                              hipStream_t stream) {
  (void)in_sizes; (void)n_in; (void)ws_size; (void)out_size;
  const float* Q = (const float*)d_in[0];
  const float* K = (const float*)d_in[1];
  const float* V = (const float*)d_in[2];
  float* O = (float*)d_out;

  __bf16* Qb = (__bf16*)d_ws;   // 3 x NTOT bf16 = ~50 MB workspace
  __bf16* Kb = Qb + NTOT;
  __bf16* Vb = Kb + NTOT;

  const int cvtBlocks = (int)(NTOT / (8 * 256));
  const float scale = 0.08838834764831845f;  // 1/sqrt(128)
  cvt_bf16_kernel<<<cvtBlocks, 256, 0, stream>>>(Q, Qb, scale);
  cvt_bf16_kernel<<<cvtBlocks, 256, 0, stream>>>(K, Kb, 1.0f);
  cvt_bf16_kernel<<<cvtBlocks, 256, 0, stream>>>(V, Vb, 1.0f);

  dim3 grid(S_ / 128, B_ * H_);
  fa_fwd_cdna5_kernel<<<grid, dim3(256), 0, stream>>>(Qb, Kb, Vb, O);
}
